// SimpleNet_order_no_batch_61332132987039
// MI455X (gfx1250) — matrix operand order repaired and checked by reading, not compiled, not run on hardware
//
#include <hip/hip_runtime.h>
#include <hip/hip_bf16.h>

// ---------------------------------------------------------------------------
// Fused DenseNet-MLP forward for MI455X (gfx1250, wave32, WMMA).
// One WG per 32 batch rows; all 24 layers run in-kernel with the full f16
// activation history (out_0..out_22) resident in LDS (row stride padded to
// 272 B -> conflict-free ds_load_b128 A fragments). Non-gathered A half is
// read straight from hist with in-register ReLU (v_pk_max_num_f16); the 128
// gathered columns use a pre-decoded LDS offset table and packed b128 writes.
// Software-pipelined: fill_gather(l+1) overlaps gemm(l) (double-buffered gbuf,
// one barrier per layer). Next layer's weights prefetched (global_prefetch_b8).
// ---------------------------------------------------------------------------

typedef _Float16 half8 __attribute__((ext_vector_type(8)));
typedef _Float16 v16h  __attribute__((ext_vector_type(16)));
typedef float    v8f   __attribute__((ext_vector_type(8)));
typedef int      v4i   __attribute__((ext_vector_type(4)));

union F16Frag { v16h v; half8 h[2]; };

#define DEPTH  24
#define WIDTH  128
#define TC     128
#define BATCH  8192
#define RTILE  32            // batch rows per workgroup
#define NLVL   23            // hist levels out_0 .. out_22
#define NMID   21            // DEPTH-3 mid layers
#define RPAD   136           // padded row stride (halves): 272 B

static __device__ __forceinline__ v8f wmma16(const F16Frag& a, const F16Frag& b, v8f c) {
  // (neg_a, A, neg_b, B, c_mod, C, reuse_a, reuse_b)
  return __builtin_amdgcn_wmma_f32_16x16x32_f16(false, a.v, false, b.v, (short)0, c,
                                                false, false);
}

static __device__ __forceinline__ v16h relu16(v16h a) {
  const v16h z = {};
  return __builtin_elementwise_max(a, z);    // -> v_pk_max_num_f16
}

// One 16-col output tile (col tile = wave id) over 2 row tiles.
// K = 0..127 from Ah (hist[l+1], optional in-register ReLU);
// K = 128..255 from Ag (gbuf, ReLU'd at fill) when GATHER.
// B from global BT[n][bstride] (N-major f16). Output raw+bias into dst (hist).
template<bool GATHER, bool DORELU>
static __device__ __forceinline__ void gemm_tiles(
    int wave, int lane,
    const _Float16* Ah, const _Float16* Ag,
    const _Float16* __restrict__ BT, int bstride,
    const float*    __restrict__ bias,
    _Float16* dst)
{
  const int nloc = lane & 15;
  const int hi   = lane >> 4;            // lane half selects K sub-chunk
  const int n    = wave * 16 + nloc;     // global output column
  v8f acc0 = {};                         // rows 0..15
  v8f acc1 = {};                         // rows 16..31
  const _Float16* bp  = BT + n * bstride + hi * 8;
  const _Float16* a0p = Ah + nloc * RPAD + hi * 8;
  const _Float16* a1p = Ah + (16 + nloc) * RPAD + hi * 8;
  #pragma unroll
  for (int k0 = 0; k0 < 128; k0 += 32) {
    F16Frag b, a0, a1;
    b.h[0]  = *(const half8*)(bp + k0);
    b.h[1]  = *(const half8*)(bp + k0 + 16);
    a0.h[0] = *(const half8*)(a0p + k0);
    a0.h[1] = *(const half8*)(a0p + k0 + 16);
    a1.h[0] = *(const half8*)(a1p + k0);
    a1.h[1] = *(const half8*)(a1p + k0 + 16);
    if (DORELU) { a0.v = relu16(a0.v); a1.v = relu16(a1.v); }
    acc0 = wmma16(a0, b, acc0);
    acc1 = wmma16(a1, b, acc1);
  }
  if (GATHER) {
    const _Float16* g0p = Ag + nloc * RPAD + hi * 8;
    const _Float16* g1p = Ag + (16 + nloc) * RPAD + hi * 8;
    #pragma unroll
    for (int k0 = 0; k0 < 128; k0 += 32) {
      F16Frag b, a0, a1;
      b.h[0]  = *(const half8*)(bp + 128 + k0);
      b.h[1]  = *(const half8*)(bp + 128 + k0 + 16);
      a0.h[0] = *(const half8*)(g0p + k0);
      a0.h[1] = *(const half8*)(g0p + k0 + 16);
      a1.h[0] = *(const half8*)(g1p + k0);
      a1.h[1] = *(const half8*)(g1p + k0 + 16);
      acc0 = wmma16(a0, b, acc0);
      acc1 = wmma16(a1, b, acc1);
    }
  }
  const float bb = bias[n];
  #pragma unroll
  for (int j = 0; j < 8; ++j) {
    const int m = hi * 8 + j;            // C/D layout: lanes 0-15 -> M=j, 16-31 -> M=8+j
    dst[m * RPAD + n]        = (_Float16)(acc0[j] + bb);
    dst[(16 + m) * RPAD + n] = (_Float16)(acc1[j] + bb);
  }
}

// Gather 128 permuted columns for layer l into gb (row-major, RPAD stride)
// using the pre-decoded LDS offset table. Thread -> (row, 16 contiguous slots):
// offsets load as 4x ds_load_b128, values packed, ReLU on packed half8,
// written as 2x ds_store_b128.
template<bool DORELU>
static __device__ __forceinline__ void fill_gather(
    int tid, int l,
    const int* goff, const _Float16* hist, _Float16* gb)
{
  const int r     = tid >> 3;            // 0..31
  const int t0    = (tid & 7) * 16;      // column group
  const int rterm = r * RPAD;
  const v4i* op = (const v4i*)(goff + l * TC + t0);
  half8 pk[2];
  #pragma unroll
  for (int q = 0; q < 4; ++q) {
    const v4i O = op[q];
    #pragma unroll
    for (int e = 0; e < 4; ++e) {
      pk[q >> 1][(q & 1) * 4 + e] = hist[O[e] + rterm];
    }
  }
  if (DORELU) {
    const half8 z = {};
    pk[0] = __builtin_elementwise_max(pk[0], z);
    pk[1] = __builtin_elementwise_max(pk[1], z);
  }
  half8* dp = (half8*)(gb + r * RPAD + t0);
  dp[0] = pk[0];
  dp[1] = pk[1];
}

// ---------------------------------------------------------------------------
// Kernel 0: convert weights to N-major f16 once per launch (deterministic).
//   W1T   [128][128]     : W1T[n][k]      = W1[k][n]
//   WmidT [21][128][256] : WmidT[l][n][k] = Wmid[l][k][n]
//   WlastT[16][256]      : cols 2..15 zero-padded -> last layer is 1 WMMA tile
// ---------------------------------------------------------------------------
__global__ void prep_weights(const float* __restrict__ W1,
                             const float* __restrict__ Wmid,
                             const float* __restrict__ Wlast,
                             _Float16* __restrict__ W1T,
                             _Float16* __restrict__ WmidT,
                             _Float16* __restrict__ WlastT)
{
  const int i  = blockIdx.x * 256 + threadIdx.x;
  const int N1 = 128 * 128;
  const int NM = NMID * 128 * 256;
  const int NL = 16 * 256;
  if (i < N1) {
    const int n = i >> 7, k = i & 127;
    W1T[i] = (_Float16)W1[k * 128 + n];
  } else if (i < N1 + NM) {
    const int j = i - N1;
    const int l = j / (128 * 256);
    const int r = j % (128 * 256);
    const int n = r >> 8, k = r & 255;
    WmidT[j] = (_Float16)Wmid[(l * 256 + k) * 128 + n];
  } else if (i < N1 + NM + NL) {
    const int j = i - N1 - NM;
    const int n = j >> 8, k = j & 255;
    WlastT[j] = (n < 2) ? (_Float16)Wlast[k * 2 + n] : (_Float16)0.f;
  }
}

// ---------------------------------------------------------------------------
// Kernel 1: fused 24-layer forward, one workgroup per 32 batch rows.
// ---------------------------------------------------------------------------
__global__ __launch_bounds__(256) void fused_simplenet(
    const float* __restrict__ x,     const int*   __restrict__ links,
    const float* __restrict__ W0,    const float* __restrict__ b0,
    const float* __restrict__ b1,    const float* __restrict__ bmid,
    const float* __restrict__ blast,
    const _Float16* __restrict__ W1T,
    const _Float16* __restrict__ WmidT,
    const _Float16* __restrict__ WlastT,
    float* __restrict__ out)
{
  __shared__ __align__(16) _Float16 hist[NLVL * RTILE * RPAD];   // ~196 KB
  __shared__ __align__(16) _Float16 gbuf[2][RTILE * RPAD];       // ~17 KB
  __shared__ __align__(16) int      goff[(NMID + 1) * TC];       // ~11 KB
  __shared__ float xs[RTILE][2];

  const int tid  = threadIdx.x;
  const int lane = tid & 31;
  const int wave = tid >> 5;
  const int row0 = blockIdx.x * RTILE;

  // stage x + decode the gather-offset table once
  if (tid < RTILE * 2) ((float*)xs)[tid] = x[row0 * 2 + tid];
  for (int i = tid; i < (NMID + 1) * TC; i += 256) {
    const int v = links[i];
    const int l = i >> 7;
    goff[i] = ((l - (v >> 7)) * RTILE) * RPAD + (v & 127);
  }
  __syncthreads();

  // ---- layer 0: out0 = relu(x @ W0 + b0), K=2 -> VALU ----
  {
    const int c  = tid & 127;
    const int rb = (tid >> 7) * 16;
    const float w00 = W0[c], w01 = W0[WIDTH + c], bb = b0[c];
    #pragma unroll
    for (int i = 0; i < 16; ++i) {
      const int r = rb + i;
      const float v = xs[r][0] * w00 + xs[r][1] * w01 + bb;
      hist[r * RPAD + c] = (_Float16)(v > 0.f ? v : 0.f);
    }
    // warm caches with the first mid-layer weight block
    const char* nw = (const char*)WmidT;
    __builtin_prefetch(nw + tid * 256, 0, 1);
    __builtin_prefetch(nw + tid * 256 + 128, 0, 1);
  }
  __syncthreads();

  // ---- layer 1 gemm (K=128; relu on out0 idempotent), overlap fill(0) ----
  gemm_tiles<false, true>(wave, lane, hist /*level 0*/, nullptr,
                          W1T, 128, b1, hist + 1 * RTILE * RPAD);
  fill_gather<true>(tid, 0, goff, hist, gbuf[0]);   // reads hist[0] only
  __syncthreads();

  // ---- mid layers l = 0..20, pipelined: gemm(l) || fill(l+1) ----
  for (int l = 0; l < NMID; ++l) {
    gemm_tiles<true, true>(wave, lane,
                           hist + (l + 1) * RTILE * RPAD, gbuf[l & 1],
                           WmidT + l * WIDTH * 256, 256, bmid + l * WIDTH,
                           hist + (l + 2) * RTILE * RPAD);
    if (l + 1 < NMID) {
      fill_gather<true>(tid, l + 1, goff, hist, gbuf[(l + 1) & 1]);
      const char* nw = (const char*)(WmidT + (l + 1) * WIDTH * 256);
      __builtin_prefetch(nw + tid * 256, 0, 1);
      __builtin_prefetch(nw + tid * 256 + 128, 0, 1);
    } else {
      fill_gather<false>(tid, NMID, goff, hist, gbuf[NMID & 1]);  // raw gather
    }
    __syncthreads();
  }

  // ---- last layer: out = [out_22 | gather_21] @ Wlast + blast  (NO relu) ----
  if (wave < 2) {                        // one 16-col (padded) tile per row tile
    const int nloc = lane & 15;
    const int hi   = lane >> 4;
    const int mrow = wave * 16;
    v8f acc = {};
    const _Float16* bp  = WlastT + nloc * 256 + hi * 8;
    const _Float16* a0p = hist + (22 * RTILE + mrow + nloc) * RPAD + hi * 8;
    const _Float16* g0p = gbuf[NMID & 1] + (mrow + nloc) * RPAD + hi * 8;
    #pragma unroll
    for (int k0 = 0; k0 < 128; k0 += 32) {
      F16Frag b, a;
      b.h[0] = *(const half8*)(bp + k0);
      b.h[1] = *(const half8*)(bp + k0 + 16);
      a.h[0] = *(const half8*)(a0p + k0);
      a.h[1] = *(const half8*)(a0p + k0 + 16);
      acc = wmma16(a, b, acc);
    }
    #pragma unroll
    for (int k0 = 0; k0 < 128; k0 += 32) {
      F16Frag b, a;
      b.h[0] = *(const half8*)(bp + 128 + k0);
      b.h[1] = *(const half8*)(bp + 128 + k0 + 16);
      a.h[0] = *(const half8*)(g0p + k0);
      a.h[1] = *(const half8*)(g0p + k0 + 16);
      acc = wmma16(a, b, acc);
    }
    if (nloc < 2) {
      const float bb = blast[nloc];
      #pragma unroll
      for (int j = 0; j < 8; ++j) {
        const int m = row0 + mrow + hi * 8 + j;
        out[m * 2 + nloc] = acc[j] + bb;
      }
    }
  }
}

// ---------------------------------------------------------------------------
extern "C" void kernel_launch(void* const* d_in, const int* in_sizes, int n_in,
                              void* d_out, int out_size, void* d_ws, size_t ws_size,
                              hipStream_t stream)
{
  const float* x     = (const float*)d_in[0];
  const int*   links = (const int*)  d_in[1];
  const float* W0    = (const float*)d_in[2];
  const float* b0    = (const float*)d_in[3];
  const float* W1    = (const float*)d_in[4];
  const float* b1    = (const float*)d_in[5];
  const float* Wmid  = (const float*)d_in[6];
  const float* bmid  = (const float*)d_in[7];
  const float* Wlast = (const float*)d_in[8];
  const float* blast = (const float*)d_in[9];
  float* out = (float*)d_out;

  _Float16* W1T    = (_Float16*)d_ws;                  //  32 KB
  _Float16* WmidT  = W1T + 128 * 128;                  // 1.31 MB
  _Float16* WlastT = WmidT + NMID * 128 * 256;         //   8 KB

  const int total = 128 * 128 + NMID * 128 * 256 + 16 * 256;
  prep_weights<<<(total + 255) / 256, 256, 0, stream>>>(W1, Wmid, Wlast,
                                                        W1T, WmidT, WlastT);
  fused_simplenet<<<BATCH / RTILE, 256, 0, stream>>>(x, links, W0, b0, b1, bmid,
                                                     blast, W1T, WmidT, WlastT, out);
}
